// CfCCell_27848567947563
// MI455X (gfx1250) — matrix operand order repaired and checked by reading, not compiled, not run on hardware
//
#include <hip/hip_runtime.h>
#include <cstdint>

typedef __attribute__((ext_vector_type(16))) __bf16 v16bf;
typedef __attribute__((ext_vector_type(8)))  float  v8f;

#define B_SZ   8192
#define IN_SZ  256
#define H_SZ   512
#define K_SZ   (IN_SZ + H_SZ)   // 768

#define BM 128
#define BN 64
#define BK 32
#define NTHREADS 256
#define LDSTR 40                 // padded LDS row stride (bf16 elems): 80B, 16B aligned
#define KSTEPS (K_SZ / BK)       // 24

union Frag { v16bf v; uint4 q[2]; };

// Pack 4 floats -> 4 bf16 with the hardware converter (v_cvt_pk_bf16_f32),
// store as one 8-byte LDS write.
__device__ __forceinline__ void st4bf(__bf16* dst, float4 v) {
  union { __bf16 b[4]; uint2 u; } t;
  t.b[0] = static_cast<__bf16>(v.x);
  t.b[1] = static_cast<__bf16>(v.y);
  t.b[2] = static_cast<__bf16>(v.z);
  t.b[3] = static_cast<__bf16>(v.w);
  *reinterpret_cast<uint2*>(dst) = t.u;
}

// tanh via the hardware exp unit: 1 - 2/(e^(2x)+1). Saturates correctly.
__device__ __forceinline__ float fast_tanh(float x) {
  return 1.0f - 2.0f / (__expf(2.0f * x) + 1.0f);
}

__global__ __launch_bounds__(NTHREADS)
void cfc_fused(const float* __restrict__ x, const float* __restrict__ h,
               const float* __restrict__ delta_t,
               const float* __restrict__ Wb,  const float* __restrict__ bb,
               const float* __restrict__ Wgx, const float* __restrict__ bgx,
               const float* __restrict__ Wgh, const float* __restrict__ gateb,
               const float* __restrict__ logtau, float* __restrict__ out)
{
  // LDS: [A 128 rows | Wg 64 rows | Wb 64 rows] x 40 bf16, double buffered = 40960 B
  __shared__ __bf16 lds[2][(BM + 2 * BN) * LDSTR];

  const int t   = threadIdx.x;
  const int bn0 = blockIdx.x * BN;
  const int bm0 = blockIdx.y * BM;

  const int lane   = t & 31;
  const int wave   = t >> 5;            // 0..7
  const int wave_m = wave >> 1;         // 0..3 : 32-row slabs
  const int wave_n = wave & 1;          // 0..1 : 32-col slabs
  const int lrow   = lane & 15;
  const int a_c0   = (lane < 16) ? 0 : 16;   // A frag K base within BK
  const int b_c0   = (lane < 16) ? 0 : 16;   // B frag K base within BK

  v8f cg[2][2], cf[2][2];
  #pragma unroll
  for (int i = 0; i < 2; ++i)
    #pragma unroll
    for (int j = 0; j < 2; ++j) { v8f z = {}; cg[i][j] = z; cf[i][j] = z; }

  auto load_stage = [&](int s, int kt) {
    const int k0 = kt * BK;
    const float* Asrc; const float* Gsrc; int lda, ldg, ka;
    if (k0 < IN_SZ) { Asrc = x; lda = IN_SZ; ka = k0;          Gsrc = Wgx; ldg = IN_SZ; }
    else            { Asrc = h; lda = H_SZ;  ka = k0 - IN_SZ;  Gsrc = Wgh; ldg = H_SZ;  }
    __bf16* Abuf = &lds[s][0];
    __bf16* Gbuf = &lds[s][BM * LDSTR];
    __bf16* Fbuf = &lds[s][(BM + BN) * LDSTR];
    #pragma unroll
    for (int it = 0; it < 4; ++it) {               // A: 128x32 = 1024 float4
      int idx = t + it * NTHREADS;
      int row = idx >> 3, c = idx & 7;
      float4 v = *reinterpret_cast<const float4*>(Asrc + (size_t)(bm0 + row) * lda + ka + c * 4);
      st4bf(Abuf + row * LDSTR + c * 4, v);
    }
    #pragma unroll
    for (int it = 0; it < 2; ++it) {               // Wg & Wb: 64x32 = 512 float4 each
      int idx = t + it * NTHREADS;
      int row = idx >> 3, c = idx & 7;
      float4 vg = *reinterpret_cast<const float4*>(Gsrc + (size_t)(bn0 + row) * ldg + ka + c * 4);
      st4bf(Gbuf + row * LDSTR + c * 4, vg);
      float4 vf = *reinterpret_cast<const float4*>(Wb + (size_t)(bn0 + row) * K_SZ + k0 + c * 4);
      st4bf(Fbuf + row * LDSTR + c * 4, vf);
    }
  };

  load_stage(0, 0);
  __syncthreads();

  for (int kt = 0; kt < KSTEPS; ++kt) {
    const int cur = kt & 1;
    if (kt + 1 < KSTEPS) load_stage(cur ^ 1, kt + 1);   // prefetch next stage

    const __bf16* Abuf = &lds[cur][0];
    const __bf16* Gbuf = &lds[cur][BM * LDSTR];
    const __bf16* Fbuf = &lds[cur][(BM + BN) * LDSTR];

    Frag a[2], bg[2], bfm[2];
    #pragma unroll
    for (int tm = 0; tm < 2; ++tm) {
      const __bf16* p = Abuf + (wave_m * 32 + tm * 16 + lrow) * LDSTR;
      a[tm].q[0] = *reinterpret_cast<const uint4*>(p + a_c0);
      a[tm].q[1] = *reinterpret_cast<const uint4*>(p + a_c0 + 8);
    }
    #pragma unroll
    for (int tn = 0; tn < 2; ++tn) {
      const __bf16* pg = Gbuf + (wave_n * 32 + tn * 16 + lrow) * LDSTR + b_c0;
      bg[tn].q[0]  = *reinterpret_cast<const uint4*>(pg);
      bg[tn].q[1]  = *reinterpret_cast<const uint4*>(pg + 8);
      const __bf16* pf = Fbuf + (wave_n * 32 + tn * 16 + lrow) * LDSTR + b_c0;
      bfm[tn].q[0] = *reinterpret_cast<const uint4*>(pf);
      bfm[tn].q[1] = *reinterpret_cast<const uint4*>(pf + 8);
    }

    #pragma unroll
    for (int tm = 0; tm < 2; ++tm)
      #pragma unroll
      for (int tn = 0; tn < 2; ++tn) {
        cg[tm][tn] = __builtin_amdgcn_wmma_f32_16x16x32_bf16(
            false, a[tm].v, false, bg[tn].v,  (short)0, cg[tm][tn], false, false);
        cf[tm][tn] = __builtin_amdgcn_wmma_f32_16x16x32_bf16(
            false, a[tm].v, false, bfm[tn].v, (short)0, cf[tm][tn], false, false);
      }
    __syncthreads();
  }

  // Epilogue: g=sigmoid(cg+bias), f=tanh(cf+bias), decay blend
  #pragma unroll
  for (int tm = 0; tm < 2; ++tm) {
    #pragma unroll
    for (int tn = 0; tn < 2; ++tn) {
      const int n   = bn0 + wave_n * 32 + tn * 16 + lrow;
      const float gb = bgx[n] + gateb[n];
      const float fb = bb[n];
      const float lt = logtau[n];
      const float sp = fmaxf(lt, 0.0f) + log1pf(__expf(-fabsf(lt)));  // softplus
      const int mbase = bm0 + wave_m * 32 + tm * 16 + ((lane < 16) ? 0 : 8);
      #pragma unroll
      for (int i = 0; i < 8; ++i) {
        const int m = mbase + i;
        const float g   = 1.0f / (1.0f + __expf(-(cg[tm][tn][i] + gb)));
        const float f   = fast_tanh(cf[tm][tn][i] + fb);
        const float tau = sp + fabsf(g);
        const float dcy = __expf(-delta_t[m] * tau);
        const float hv  = h[(size_t)m * H_SZ + n];
        out[(size_t)m * H_SZ + n] = dcy * hv + (1.0f - dcy) * f;
      }
    }
  }
}

extern "C" void kernel_launch(void* const* d_in, const int* in_sizes, int n_in,
                              void* d_out, int out_size, void* d_ws, size_t ws_size,
                              hipStream_t stream) {
  (void)in_sizes; (void)n_in; (void)d_ws; (void)ws_size; (void)out_size;
  const float* x      = (const float*)d_in[0];
  const float* h      = (const float*)d_in[1];
  const float* dt     = (const float*)d_in[2];
  const float* Wb     = (const float*)d_in[3];
  const float* bb     = (const float*)d_in[4];
  const float* Wgx    = (const float*)d_in[5];
  const float* bgx    = (const float*)d_in[6];
  const float* Wgh    = (const float*)d_in[7];
  const float* gateb  = (const float*)d_in[8];
  const float* logtau = (const float*)d_in[9];
  float* out = (float*)d_out;

  dim3 grid(H_SZ / BN, B_SZ / BM);   // (8, 64) = 512 workgroups
  dim3 block(NTHREADS);
  hipLaunchKernelGGL(cfc_fused, grid, block, 0, stream,
                     x, h, dt, Wb, bb, Wgx, bgx, Wgh, gateb, logtau, out);
}
